// SingleVisLoss_13743895347724
// MI455X (gfx1250) — matrix operand order repaired and checked by reading, not compiled, not run on hardware
//
#include <hip/hip_runtime.h>
#include <hip/hip_bf16.h>
#include <math.h>

#define B_N   4096
#define D_DIM 512

typedef __attribute__((ext_vector_type(16))) __bf16 v16bf;
typedef __attribute__((ext_vector_type(8)))  __bf16 v8bf;
typedef __attribute__((ext_vector_type(8)))  float  v8f;

// ---------------- block reduction helper (deterministic shuffle tree) ----
__device__ __forceinline__ float blockReduceSum(float v) {
    __shared__ float red[32];
    #pragma unroll
    for (int m = 16; m; m >>= 1) v += __shfl_xor(v, m, 32);
    int lane = threadIdx.x & 31;
    int w    = threadIdx.x >> 5;
    if (lane == 0) red[w] = v;
    __syncthreads();
    int nw = ((int)blockDim.x + 31) >> 5;
    v = (threadIdx.x < (unsigned)nw) ? red[threadIdx.x] : 0.0f;
    if (w == 0) {
        #pragma unroll
        for (int m = 16; m; m >>= 1) v += __shfl_xor(v, m, 32);
    }
    return v; // valid on thread 0
}

// ---------------- init per-row stats ----------------
__global__ void init_ws_kernel(int* kcnt, int* rep, int* dminb, int* dmaxb) {
    int i = blockIdx.x * blockDim.x + threadIdx.x;
    if (i < B_N) {
        kcnt[i]  = 0;
        rep[i]   = 0x7FFFFFFF;
        dminb[i] = 0x7F800000;  // +inf bits (distances are >= 0)
        dmaxb[i] = 0;           // 0.0f bits
    }
}

// ------- row norms of bf16-rounded edge_from (+ optional bf16 pack) ------
template <bool PACKED>
__global__ void row_norms_kernel(const float* __restrict__ ef, float* __restrict__ nrm,
                                 __bf16* __restrict__ efb) {
    int wave = threadIdx.x >> 5;
    int lane = threadIdx.x & 31;
    int row  = blockIdx.x * ((int)blockDim.x >> 5) + wave;
    if (row >= B_N) return;
    const float* p = ef + (size_t)row * D_DIM + lane * 16;
    float s = 0.0f;
    v16bf vb;
    #pragma unroll
    for (int e = 0; e < 16; ++e) {
        __bf16 xb = (__bf16)p[e];       // same rounding the WMMA operands see
        vb[e] = xb;
        float xf = (float)xb;
        s += xf * xf;
    }
    if (PACKED) {
        *(v16bf*)(efb + (size_t)row * D_DIM + lane * 16) = vb;  // 32B aligned
    }
    #pragma unroll
    for (int m = 16; m; m >>= 1) s += __shfl_xor(s, m, 32);
    if (lane == 0) nrm[row] = s;
}

// ---------------- umap partials: mean log1p(||e_to - e_from||^2) ---------
__global__ void umap_kernel(const float* __restrict__ et, const float* __restrict__ efm,
                            float* __restrict__ part_u) {
    int i = blockIdx.x * blockDim.x + threadIdx.x;
    float v = 0.0f;
    if (i < B_N) {
        float dx = et[2 * i]     - efm[2 * i];
        float dy = et[2 * i + 1] - efm[2 * i + 1];
        v = log1pf(dx * dx + dy * dy);
    }
    v = blockReduceSum(v);
    if (threadIdx.x == 0) part_u[blockIdx.x] = v;
}

// ---------------- recon partials: sum of squared diffs (both pairs) ------
__global__ void recon_kernel(const float* __restrict__ et, const float* __restrict__ ef,
                             const float* __restrict__ rt, const float* __restrict__ rf,
                             float* __restrict__ part_r) {
    const int N4 = B_N * D_DIM / 4;
    float s = 0.0f;
    for (int i = blockIdx.x * blockDim.x + threadIdx.x; i < N4;
         i += gridDim.x * blockDim.x) {
        float4 a = ((const float4*)et)[i];
        float4 b = ((const float4*)rt)[i];
        float4 c = ((const float4*)ef)[i];
        float4 d = ((const float4*)rf)[i];
        float t;
        t = b.x - a.x; s += t * t;  t = b.y - a.y; s += t * t;
        t = b.z - a.z; s += t * t;  t = b.w - a.w; s += t * t;
        t = d.x - c.x; s += t * t;  t = d.y - c.y; s += t * t;
        t = d.z - c.z; s += t * t;  t = d.w - c.w; s += t * t;
    }
    s = blockReduceSum(s);
    if (threadIdx.x == 0) part_r[blockIdx.x] = s;
}

// ---------------- WMMA Gram + group-mask + per-row min/max stats ---------
// One wave computes a 32x64 strip: 2 A-tiles x 4 B-tiles = 8 WMMA per k-step,
// all 12 b128 fragment loads issued before the first WMMA of the step so the
// scheduler can wait incrementally and overlap loads with matrix ops.
// Upper triangle only; sub-diagonal sub-tiles masked, tj>ti mirrored.
// A frag (16-bit 16x32): lanes 0-15: M=l15, elems[0..15]=K+0..15;
//                        lanes 16-31: K+16..31.
// B frag (32x16): lanes 0-15: N=l15, K+0..15; lanes 16-31: N=l15, K+16..31.
// C layout: vgpr r -> M = r + 8*half, N = l15.
template <bool PACKED>
__global__ __launch_bounds__(32)
void gram_mask_wmma(const float* __restrict__ ef, const __bf16* __restrict__ efb,
                    const float* __restrict__ embt, const float* __restrict__ nrm,
                    int* kcnt, int* rep, int* dminb, int* dmaxb) {
    int tj4 = blockIdx.x;              // 64-column group (4 tiles)
    int ti2 = blockIdx.y;              // 32-row group (2 tiles)
    if (tj4 * 4 + 3 < ti2 * 2) return; // group entirely below diagonal
    int lane = threadIdx.x & 31;
    int hl   = lane >> 4;
    int l15  = lane & 15;
    int i0 = ti2 * 32, j0 = tj4 * 64;

    v8f acc[8];
    #pragma unroll
    for (int t = 0; t < 8; ++t) acc[t] = (v8f){};

    const float*  paf[2]; const __bf16* pab[2];
    const float*  pbf[4]; const __bf16* pbb[4];
    #pragma unroll
    for (int p = 0; p < 2; ++p) {
        size_t off = (size_t)(i0 + p * 16 + l15) * D_DIM + (hl ? 16 : 0);
        paf[p] = ef + off;  pab[p] = efb + off;
        if (PACKED) __builtin_prefetch((const void*)pab[p], 0, 0);
        else        __builtin_prefetch((const void*)paf[p], 0, 0);
    }
    #pragma unroll
    for (int q = 0; q < 4; ++q) {
        size_t off = (size_t)(j0 + q * 16 + l15) * D_DIM + (hl ? 16 : 0);
        pbf[q] = ef + off;  pbb[q] = efb + off;
        if (PACKED) __builtin_prefetch((const void*)pbb[q], 0, 0);
        else        __builtin_prefetch((const void*)pbf[q], 0, 0);
    }

    #pragma unroll 2
    for (int kk = 0; kk < D_DIM; kk += 32) {
        v16bf av[2], bv[4];
        // ---- issue ALL fragment loads for this k-step first ----
        #pragma unroll
        for (int p = 0; p < 2; ++p) {
            if (PACKED) {
                union { v16bf v; v8bf h[2]; } ua;
                ua.h[0] = *(const v8bf*)(pab[p] + kk);
                ua.h[1] = *(const v8bf*)(pab[p] + kk + 8);
                av[p] = ua.v;
            } else {
                v8f alo = *(const v8f*)(paf[p] + kk);
                v8f ahi = *(const v8f*)(paf[p] + kk + 8);
                #pragma unroll
                for (int e = 0; e < 8; ++e) {
                    av[p][e] = (__bf16)alo[e]; av[p][e + 8] = (__bf16)ahi[e];
                }
            }
        }
        #pragma unroll
        for (int q = 0; q < 4; ++q) {
            if (PACKED) {
                bv[q] = *(const v16bf*)(pbb[q] + kk);
            } else {
                v8f blo = *(const v8f*)(pbf[q] + kk);
                v8f bhi = *(const v8f*)(pbf[q] + kk + 8);
                #pragma unroll
                for (int e = 0; e < 8; ++e) {
                    bv[q][e] = (__bf16)blo[e]; bv[q][e + 8] = (__bf16)bhi[e];
                }
            }
        }
        // ---- 8 WMMAs on the staged fragments ----
        #pragma unroll
        for (int p = 0; p < 2; ++p)
            #pragma unroll
            for (int q = 0; q < 4; ++q)
                acc[p * 4 + q] = __builtin_amdgcn_wmma_f32_16x16x32_bf16(
                        false, av[p], false, bv[q], (short)0, acc[p * 4 + q],
                        false, false);
    }

    // ---- epilogue: threshold mask + per-row member stats ----
    #pragma unroll
    for (int q = 0; q < 4; ++q) {
        int   tj  = tj4 * 4 + q;
        int   j   = j0 + q * 16 + l15;
        float nj  = nrm[j];
        float ejx = embt[2 * j];
        float ejy = embt[2 * j + 1];
        #pragma unroll
        for (int p = 0; p < 2; ++p) {
            int ti = ti2 * 2 + p;
            if (tj < ti) continue;       // sub-diagonal sub-tile
            v8f c = acc[p * 4 + q];
            #pragma unroll
            for (int r = 0; r < 8; ++r) {
                int   i   = i0 + p * 16 + r + (hl ? 8 : 0);
                float d2f = nrm[i] + nj - 2.0f * c[r];
                if (d2f < 1e-3f) {       // same group
                    float dx = embt[2 * i]     - ejx;
                    float dy = embt[2 * i + 1] - ejy;
                    float d  = sqrtf(fmaxf(dx * dx + dy * dy, 1e-12f));
                    int   db = __float_as_int(d);  // >=0: int order == float order
                    atomicAdd(&kcnt[i], 1);
                    atomicMin(&rep[i], j);
                    atomicMin(&dminb[i], db);
                    atomicMax(&dmaxb[i], db);
                    if (ti != tj) {      // mirror to lower triangle
                        atomicAdd(&kcnt[j], 1);
                        atomicMin(&rep[j], i);
                        atomicMin(&dminb[j], db);
                        atomicMax(&dmaxb[j], db);
                    }
                }
            }
        }
    }
}

// ---------------- finalize: deterministic segment reduction --------------
__global__ __launch_bounds__(1024)
void finalize_kernel(const int* __restrict__ kcnt, const int* __restrict__ rep,
                     const int* __restrict__ dminb, const int* __restrict__ dmaxb,
                     const float* __restrict__ part_u, const float* __restrict__ part_r,
                     int n_part_u, int n_part_r, float* __restrict__ out) {
    __shared__ float rrow[B_N];
    __shared__ int   rrep[B_N];
    for (int i = threadIdx.x; i < B_N; i += blockDim.x) {
        int   k    = kcnt[i];
        float dmax = __int_as_float(dmaxb[i]);
        float dmin = __int_as_float(dminb[i]);
        int   den  = k - 1; if (den < 1) den = 1;
        rrow[i] = (dmax - dmin) / (float)den;   // telescoped sorted-diff sum
        rrep[i] = rep[i];
    }
    __syncthreads();
    float s = 0.0f, nv = 0.0f;
    for (int g = threadIdx.x; g < B_N; g += blockDim.x) {
        float sum = 0.0f; int cnt = 0;
        for (int i = 0; i < B_N; ++i) {
            if (rrep[i] == g) { sum += rrow[i]; ++cnt; }
        }
        if (cnt >= 2) { s += sum / (float)cnt; nv += 1.0f; }
    }
    s = blockReduceSum(s);
    __syncthreads();
    nv = blockReduceSum(nv);
    if (threadIdx.x == 0) {
        float um = 0.0f;
        for (int i = 0; i < n_part_u; ++i) um += part_u[i];
        float rc = 0.0f;
        for (int i = 0; i < n_part_r; ++i) rc += part_r[i];
        um /= (float)B_N;
        rc /= (float)(B_N * D_DIM);
        float rank = s / fmaxf(nv, 1.0f);
        out[0] = um;
        out[1] = rc;
        out[2] = rank;
        out[3] = um + 1.0f * rc + rank;    // LAMBD = 1.0
    }
}

extern "C" void kernel_launch(void* const* d_in, const int* in_sizes, int n_in,
                              void* d_out, int out_size, void* d_ws, size_t ws_size,
                              hipStream_t stream) {
    (void)in_sizes; (void)n_in; (void)out_size;
    const float* edge_to    = (const float*)d_in[0];
    const float* edge_from  = (const float*)d_in[1];
    const float* emb_to     = (const float*)d_in[2];
    const float* emb_from   = (const float*)d_in[3];
    const float* recon_to   = (const float*)d_in[4];
    const float* recon_from = (const float*)d_in[5];
    float* out = (float*)d_out;

    const int UMAP_BLOCKS  = (B_N + 255) / 256;   // 16
    const int RECON_BLOCKS = 256;

    const size_t BF16_BYTES  = (size_t)B_N * D_DIM * sizeof(__bf16);  // 4 MB
    const size_t STATS_BYTES = (size_t)5 * B_N * 4
                             + (size_t)(UMAP_BLOCKS + RECON_BLOCKS) * 4;
    // Deterministic per-run: ws_size is fixed by the harness.
    const bool packed = ws_size >= BF16_BYTES + STATS_BYTES;

    char*   base = (char*)d_ws;
    __bf16* efb  = packed ? (__bf16*)base : nullptr;  // 256B-aligned -> 32B ok
    char*   sb   = base + (packed ? BF16_BYTES : 0);

    float* nrm    = (float*)sb;  sb += B_N * sizeof(float);
    int*   kcnt   = (int*)sb;    sb += B_N * sizeof(int);
    int*   rep    = (int*)sb;    sb += B_N * sizeof(int);
    int*   dminb  = (int*)sb;    sb += B_N * sizeof(int);
    int*   dmaxb  = (int*)sb;    sb += B_N * sizeof(int);
    float* part_u = (float*)sb;  sb += UMAP_BLOCKS * sizeof(float);
    float* part_r = (float*)sb;

    init_ws_kernel<<<(B_N + 255) / 256, 256, 0, stream>>>(kcnt, rep, dminb, dmaxb);
    if (packed)
        row_norms_kernel<true><<<B_N / 8, 256, 0, stream>>>(edge_from, nrm, efb);
    else
        row_norms_kernel<false><<<B_N / 8, 256, 0, stream>>>(edge_from, nrm, nullptr);
    umap_kernel<<<UMAP_BLOCKS, 256, 0, stream>>>(emb_to, emb_from, part_u);
    recon_kernel<<<RECON_BLOCKS, 256, 0, stream>>>(edge_to, edge_from, recon_to,
                                                   recon_from, part_r);
    dim3 g(B_N / 64, B_N / 32);   // (tj groups of 4 tiles, ti groups of 2 tiles)
    if (packed)
        gram_mask_wmma<true><<<g, 32, 0, stream>>>(edge_from, efb, emb_to, nrm,
                                                   kcnt, rep, dminb, dmaxb);
    else
        gram_mask_wmma<false><<<g, 32, 0, stream>>>(edge_from, nullptr, emb_to, nrm,
                                                    kcnt, rep, dminb, dmaxb);
    finalize_kernel<<<1, 1024, 0, stream>>>(kcnt, rep, dminb, dmaxb,
                                            part_u, part_r,
                                            UMAP_BLOCKS, RECON_BLOCKS, out);
}
